// myModel_81312320848265
// MI455X (gfx1250) — matrix operand order repaired and checked by reading, not compiled, not run on hardware
//
#include <hip/hip_runtime.h>
#include <hip/hip_bf16.h>
#include <cmath>

// ---------------------------------------------------------------------------
// Self-attention + Gram matrix for MI455X (gfx1250), bf16 WMMA path with
// double-buffered async global->LDS staging (GLOBAL_LOAD_ASYNC_TO_LDS_B128).
//   x:[4096,2348] fp32 -> q,k,v = x@W^T + b -> S = softmax(q k^T / sqrt(D))
//   Y = S@v -> out = Y@Y^T  (fp32 [4096,4096])
// ---------------------------------------------------------------------------

typedef __attribute__((ext_vector_type(16))) __bf16 bf16x16;
typedef __attribute__((ext_vector_type(8)))  __bf16 bf16x8;
typedef __attribute__((ext_vector_type(8)))  float  f32x8;
typedef __attribute__((ext_vector_type(4)))  int    i32x4;

#define NS 4096          // rows (shots)
#define DF 2348          // true feature dim
#define KP 2368          // DF padded to multiple of 32 (K of projection GEMMs)
#define DP 2432          // DF padded to multiple of 128 (cols of q/k/v/Y)

#define AS1 __attribute__((address_space(1)))
#define AS3 __attribute__((address_space(3)))

typedef AS1 i32x4 g_i32x4;   // global-AS 16B chunk (source, non-const per builtin)
typedef AS3 i32x4 l_i32x4;   // LDS-AS 16B chunk (destination)

#if defined(__HIP_DEVICE_COMPILE__) && \
    __has_builtin(__builtin_amdgcn_global_load_async_to_lds_b128)
#define USE_ASYNC_LDS 1
#else
#define USE_ASYNC_LDS 0
#endif

__device__ __forceinline__ void wait_asynccnt0() {
#if defined(__HIP_DEVICE_COMPILE__)
#if __has_builtin(__builtin_amdgcn_s_wait_asynccnt)
  __builtin_amdgcn_s_wait_asynccnt(0);
#else
  asm volatile("s_wait_asynccnt 0x0" ::: "memory");
#endif
#endif
}

// ------------------------------ packing ------------------------------------
__global__ __launch_bounds__(256) void pack_x_bf16(const float* __restrict__ x,
                                                   __bf16* __restrict__ xb) {
  size_t i = (size_t)blockIdx.x * 256u + threadIdx.x;
  if (i >= (size_t)NS * KP) return;
  int r = (int)(i / KP), c = (int)(i % KP);
  float v = (c < DF) ? x[(size_t)r * DF + c] : 0.0f;
  xb[i] = (__bf16)v;
}

__global__ __launch_bounds__(256) void pack_w_bf16(const float* __restrict__ w,
                                                   __bf16* __restrict__ wb) {
  size_t i = (size_t)blockIdx.x * 256u + threadIdx.x;
  if (i >= (size_t)DP * KP) return;
  int r = (int)(i / KP), c = (int)(i % KP);
  float v = (r < DF && c < DF) ? w[(size_t)r * DF + c] : 0.0f;
  wb[i] = (__bf16)v;
}

// ------------------------------ WMMA GEMM ----------------------------------
// C[M x Ncols] = A[M x K] * op(B) (+ bias) * scale
//   BT=true : B is [Ncols x K] row-major (C = A * B^T)   -- "NT"
//   BT=false: B is [K x Ncols] row-major (C = A * B)     -- "NN"
// Block tile 128x128, 256 threads = 8 wave32; wave tile 32(M) x 64(N).
template <bool BT, bool OUT_F32, bool HAS_BIAS>
__global__ __launch_bounds__(256)
void gemm_wmma_bf16(const __bf16* __restrict__ A, int lda,
                    const __bf16* __restrict__ B, int ldb,
                    void* __restrict__ Cout, int ldc,
                    const float* __restrict__ bias, int biasLen,
                    int K, float scale) {
  const int tid  = threadIdx.x;
  const int lane = tid & 31;
  const int wave = tid >> 5;
  const int wm   = wave >> 1;       // 0..3  (M sub-tile, 32 rows each)
  const int wn   = wave & 1;        // 0..1  (N sub-tile, 64 cols each)
  const int row0 = blockIdx.y * 128;
  const int col0 = blockIdx.x * 128;

  f32x8 acc[2][4];
#pragma unroll
  for (int mi = 0; mi < 2; ++mi)
#pragma unroll
    for (int ni = 0; ni < 4; ++ni)
      acc[mi][ni] = (f32x8){0.f, 0.f, 0.f, 0.f, 0.f, 0.f, 0.f, 0.f};

  // per-lane fragment geometry (CDNA5 16x16x32 bf16 layouts)
  const int kbA = (lane < 16) ? 0 : 8;    // A: half-waves split K 0..7 / 8..15 (+16)
  const int kgB = (lane < 16) ? 0 : 8;    // B: half-waves split K 0..7 / 8..15 (+16)
  const int l16 = lane & 15;

#if USE_ASYNC_LDS
  // ---- double-buffered async pipeline -------------------------------------
  __shared__ __align__(16) __bf16 lA[2][128][40];   // [buf][m][k], padded rows
  __shared__ __align__(16) __bf16 lB[2][128][40];   // [buf][n][k], padded rows

  auto stage = [&](int buf, int k0) {
    const __bf16* Ab = A + (size_t)row0 * lda + k0;
#pragma unroll
    for (int c = tid; c < 512; c += 256) {
      int r = c >> 2, ch = (c & 3) * 8;
      __builtin_amdgcn_global_load_async_to_lds_b128(
          (g_i32x4*)(Ab + (size_t)r * lda + ch),
          (l_i32x4*)&lA[buf][r][ch], 0, 0);
    }
    if (BT) {
      const __bf16* Bb = B + (size_t)col0 * ldb + k0;
#pragma unroll
      for (int c = tid; c < 512; c += 256) {
        int r = c >> 2, ch = (c & 3) * 8;
        __builtin_amdgcn_global_load_async_to_lds_b128(
            (g_i32x4*)(Bb + (size_t)r * ldb + ch),
            (l_i32x4*)&lB[buf][r][ch], 0, 0);
      }
    } else {  // transposing copy: B is [K x Ncols]
      int kk = tid & 31, nb = tid >> 5;   // 32 k x 8 groups of 16 n
      const __bf16* p = B + (size_t)(k0 + kk) * ldb + col0 + nb * 16;
      bf16x8 t0 = *(const bf16x8*)p;
      bf16x8 t1 = *(const bf16x8*)(p + 8);
#pragma unroll
      for (int j = 0; j < 8; ++j) {
        lB[buf][nb * 16 + j][kk]     = t0[j];
        lB[buf][nb * 16 + 8 + j][kk] = t1[j];
      }
    }
  };

  const int nk = K >> 5;
  stage(0, 0);
  for (int it = 0; it < nk; ++it) {
    const int buf = it & 1;
    wait_asynccnt0();       // this wave's async DMAs for tile `it` landed
    __syncthreads();        // whole tile ready; prior reads of buf^1 done
    if (it + 1 < nk) stage(buf ^ 1, (it + 1) << 5);

    bf16x16 afr[2];
#pragma unroll
    for (int mi = 0; mi < 2; ++mi) {
      int m = wm * 32 + mi * 16 + l16;
      bf16x8 lo = *(const bf16x8*)&lA[buf][m][kbA];
      bf16x8 hi = *(const bf16x8*)&lA[buf][m][kbA + 16];
#pragma unroll
      for (int j = 0; j < 8; ++j) { afr[mi][j] = lo[j]; afr[mi][j + 8] = hi[j]; }
    }
    bf16x16 bfr[4];
#pragma unroll
    for (int ni = 0; ni < 4; ++ni) {
      int n = wn * 64 + ni * 16 + l16;
      bf16x8 lo = *(const bf16x8*)&lB[buf][n][kgB];
      bf16x8 hi = *(const bf16x8*)&lB[buf][n][kgB + 16];
#pragma unroll
      for (int j = 0; j < 8; ++j) { bfr[ni][j] = lo[j]; bfr[ni][j + 8] = hi[j]; }
    }

#pragma unroll
    for (int mi = 0; mi < 2; ++mi)
#pragma unroll
      for (int ni = 0; ni < 4; ++ni)
        acc[mi][ni] = __builtin_amdgcn_wmma_f32_16x16x32_bf16(
            false, afr[mi], false, bfr[ni], (short)0, acc[mi][ni], false, false);
  }
#else
  // ---- fallback: single-buffer synchronous staging ------------------------
  __shared__ __align__(16) __bf16 lA[128][40];
  __shared__ __align__(16) __bf16 lB[128][40];

  for (int k0 = 0; k0 < K; k0 += 32) {
    __syncthreads();
    {
      const __bf16* Ab = A + (size_t)row0 * lda + k0;
#pragma unroll
      for (int c = tid; c < 512; c += 256) {
        int r = c >> 2, ch = (c & 3) * 8;
        *(uint4*)&lA[r][ch] = *(const uint4*)(Ab + (size_t)r * lda + ch);
      }
    }
    if (BT) {
      const __bf16* Bb = B + (size_t)col0 * ldb + k0;
#pragma unroll
      for (int c = tid; c < 512; c += 256) {
        int r = c >> 2, ch = (c & 3) * 8;
        *(uint4*)&lB[r][ch] = *(const uint4*)(Bb + (size_t)r * ldb + ch);
      }
    } else {
      int kk = tid & 31, nb = tid >> 5;
      const __bf16* p = B + (size_t)(k0 + kk) * ldb + col0 + nb * 16;
      bf16x8 t0 = *(const bf16x8*)p;
      bf16x8 t1 = *(const bf16x8*)(p + 8);
#pragma unroll
      for (int j = 0; j < 8; ++j) {
        lB[nb * 16 + j][kk]     = t0[j];
        lB[nb * 16 + 8 + j][kk] = t1[j];
      }
    }
    __syncthreads();

    bf16x16 afr[2];
#pragma unroll
    for (int mi = 0; mi < 2; ++mi) {
      int m = wm * 32 + mi * 16 + l16;
      bf16x8 lo = *(const bf16x8*)&lA[m][kbA];
      bf16x8 hi = *(const bf16x8*)&lA[m][kbA + 16];
#pragma unroll
      for (int j = 0; j < 8; ++j) { afr[mi][j] = lo[j]; afr[mi][j + 8] = hi[j]; }
    }
    bf16x16 bfr[4];
#pragma unroll
    for (int ni = 0; ni < 4; ++ni) {
      int n = wn * 64 + ni * 16 + l16;
      bf16x8 lo = *(const bf16x8*)&lB[n][kgB];
      bf16x8 hi = *(const bf16x8*)&lB[n][kgB + 16];
#pragma unroll
      for (int j = 0; j < 8; ++j) { bfr[ni][j] = lo[j]; bfr[ni][j + 8] = hi[j]; }
    }

#pragma unroll
    for (int mi = 0; mi < 2; ++mi)
#pragma unroll
      for (int ni = 0; ni < 4; ++ni)
        acc[mi][ni] = __builtin_amdgcn_wmma_f32_16x16x32_bf16(
            false, afr[mi], false, bfr[ni], (short)0, acc[mi][ni], false, false);
  }
#endif

  // ---- epilogue: C/D layout -> lanes 0-15 M=r, lanes 16-31 M=8+r ----
#pragma unroll
  for (int mi = 0; mi < 2; ++mi) {
#pragma unroll
    for (int ni = 0; ni < 4; ++ni) {
      int col   = col0 + wn * 64 + ni * 16 + l16;
      int rbase = row0 + wm * 32 + mi * 16 + ((lane & 16) ? 8 : 0);
      float badd = 0.0f;
      if (HAS_BIAS) badd = (col < biasLen) ? bias[col] : 0.0f;
#pragma unroll
      for (int r = 0; r < 8; ++r) {
        float v = acc[mi][ni][r] * scale + badd;
        size_t idx = (size_t)(rbase + r) * ldc + col;
        if (OUT_F32) ((float*)Cout)[idx] = v;
        else         ((__bf16*)Cout)[idx] = (__bf16)v;
      }
    }
  }
}

// ------------------------------ softmax ------------------------------------
__global__ __launch_bounds__(256)
void softmax_rows(const float* __restrict__ S, __bf16* __restrict__ P, int n) {
  const int row = blockIdx.x;
  const float* s = S + (size_t)row * n;
  __bf16*      p = P + (size_t)row * n;
  const int tid = threadIdx.x;
  __shared__ float red[8];

  float mx = -3.4e38f;
  for (int c = tid; c < n; c += 256) mx = fmaxf(mx, s[c]);
#pragma unroll
  for (int o = 16; o > 0; o >>= 1) mx = fmaxf(mx, __shfl_xor(mx, o, 32));
  if ((tid & 31) == 0) red[tid >> 5] = mx;
  __syncthreads();
  mx = red[0];
#pragma unroll
  for (int i = 1; i < 8; ++i) mx = fmaxf(mx, red[i]);
  __syncthreads();

  float sum = 0.0f;
  for (int c = tid; c < n; c += 256) sum += __expf(s[c] - mx);
#pragma unroll
  for (int o = 16; o > 0; o >>= 1) sum += __shfl_xor(sum, o, 32);
  if ((tid & 31) == 0) red[tid >> 5] = sum;
  __syncthreads();
  sum = 0.0f;
#pragma unroll
  for (int i = 0; i < 8; ++i) sum += red[i];
  float inv = 1.0f / sum;

  for (int c = tid; c < n; c += 256) p[c] = (__bf16)(__expf(s[c] - mx) * inv);
}

// ------------------------------ driver -------------------------------------
extern "C" void kernel_launch(void* const* d_in, const int* in_sizes, int n_in,
                              void* d_out, int out_size, void* d_ws, size_t ws_size,
                              hipStream_t stream) {
  const float* x  = (const float*)d_in[0];
  const float* Wq = (const float*)d_in[3];
  const float* bq = (const float*)d_in[4];
  const float* Wk = (const float*)d_in[5];
  const float* bk = (const float*)d_in[6];
  const float* Wv = (const float*)d_in[7];
  const float* bv = (const float*)d_in[8];

  // workspace layout (all sizes multiples of 256 B)
  char* w = (char*)d_ws;
  const size_t szXB = (size_t)NS * KP * 2;   // 19,398,656
  const size_t szWB = (size_t)DP * KP * 2;   // 11,517,952
  const size_t szQB = (size_t)NS * DP * 2;   // 19,922,944
  const size_t szS  = (size_t)NS * NS * 4;   // 67,108,864
  const size_t szP  = (size_t)NS * NS * 2;   // 33,554,432
  __bf16* xb  = (__bf16*)w;                 w += szXB;
  __bf16* wqb = (__bf16*)w;                 w += szWB;
  __bf16* wkb = (__bf16*)w;                 w += szWB;
  __bf16* wvb = (__bf16*)w;                 w += szWB;
  __bf16* qb  = (__bf16*)w;                 w += szQB;
  __bf16* kb  = (__bf16*)w;                 w += szQB;
  __bf16* vb  = (__bf16*)w;                 w += szQB;
  float*  Sf  = (float*)w;                  w += szS;
  __bf16* Pb  = (__bf16*)w;                 w += szP;
  __bf16* Yb  = (__bf16*)w;                 w += szQB;

  const float scl = 1.0f / sqrtf((float)DF);

  // 1) pack to bf16 (zero-padded)
  {
    int gx = (int)(((size_t)NS * KP + 255) / 256);
    pack_x_bf16<<<gx, 256, 0, stream>>>(x, xb);
    int gw = (int)(((size_t)DP * KP + 255) / 256);
    pack_w_bf16<<<gw, 256, 0, stream>>>(Wq, wqb);
    pack_w_bf16<<<gw, 256, 0, stream>>>(Wk, wkb);
    pack_w_bf16<<<gw, 256, 0, stream>>>(Wv, wvb);
  }

  dim3 blk(256);
  dim3 gQKV(DP / 128, NS / 128);   // 19 x 32
  dim3 gNN(NS / 128, NS / 128);    // 32 x 32

  // 2) q,k,v = x @ W^T + b   (NT, bf16 out, bias)
  gemm_wmma_bf16<true, false, true><<<gQKV, blk, 0, stream>>>(
      xb, KP, wqb, KP, qb, DP, bq, DF, KP, 1.0f);
  gemm_wmma_bf16<true, false, true><<<gQKV, blk, 0, stream>>>(
      xb, KP, wkb, KP, kb, DP, bk, DF, KP, 1.0f);
  gemm_wmma_bf16<true, false, true><<<gQKV, blk, 0, stream>>>(
      xb, KP, wvb, KP, vb, DP, bv, DF, KP, 1.0f);

  // 3) S = (q @ k^T) * 1/sqrt(D)   (NT, fp32 out)
  gemm_wmma_bf16<true, true, false><<<gNN, blk, 0, stream>>>(
      qb, DP, kb, DP, Sf, NS, nullptr, 0, DP, scl);

  // 4) P = softmax_rows(S), bf16
  softmax_rows<<<NS, blk, 0, stream>>>(Sf, Pb, NS);

  // 5) Y = P @ v   (NN, bf16 out)
  gemm_wmma_bf16<false, false, false><<<gQKV, blk, 0, stream>>>(
      Pb, NS, vb, DP, Yb, DP, nullptr, 0, NS, 1.0f);

  // 6) out = Y @ Y^T   (NT, fp32 out -> d_out)
  gemm_wmma_bf16<true, true, false><<<gNN, blk, 0, stream>>>(
      Yb, DP, Yb, DP, (float*)d_out, NS, nullptr, 0, DP, 1.0f);
}
